// Head_PCL_51737176048186
// MI455X (gfx1250) — matrix operand order repaired and checked by reading, not compiled, not run on hardware
//
#include <hip/hip_runtime.h>
#include <math.h>

// ---------------- Problem constants (match reference setup_inputs) ----------
#define Bc   4
#define Cc   2048
#define Tc   2048
#define Pc   256
#define Hc   4096
#define NCc  20
#define Rc   3
#define Dc   (7 * Cc)          // 14336
#define Mc   (Bc * Pc)         // 1024
#define BCc  (Bc * Cc)         // 8192

typedef __attribute__((ext_vector_type(16))) __bf16 v16bf;
typedef __attribute__((ext_vector_type(8)))  __bf16 v8bf;
typedef __attribute__((ext_vector_type(8)))  float  v8f;

union Frag16 { v16bf v; v8bf h[2]; };

__device__ __forceinline__ unsigned short f2bf(float f) {
  union { float f; unsigned u; } x; x.f = f;
  unsigned u = x.u;
  unsigned r = (u + 0x7FFFu + ((u >> 16) & 1u)) >> 16;   // round-to-nearest-even
  return (unsigned short)r;
}

__device__ __forceinline__ float wred_sum(float v) {
  #pragma unroll
  for (int o = 16; o > 0; o >>= 1) v += __shfl_xor(v, o, 32);
  return v;
}
__device__ __forceinline__ float wred_max(float v) {
  #pragma unroll
  for (int o = 16; o > 0; o >>= 1) v = fmaxf(v, __shfl_xor(v, o, 32));
  return v;
}

// ---------------- CDNA5 async global->LDS copy (ASYNCcnt-tracked) ----------
#if defined(__has_builtin)
#  if __has_builtin(__builtin_amdgcn_global_load_async_to_lds_b128)
#    define HAVE_ASYNC_B128 1
#  endif
#  if __has_builtin(__builtin_amdgcn_s_wait_asynccnt)
#    define HAVE_WAIT_ASYNC 1
#  endif
#endif

typedef int v4i_t __attribute__((vector_size(16)));
typedef __attribute__((address_space(1))) v4i_t v4i_glob;   // global int4
typedef __attribute__((address_space(3))) v4i_t v4i_lds;    // LDS int4

__device__ __forceinline__ void async_copy16(const void* g, void* l) {
#if defined(HAVE_ASYNC_B128)
  __builtin_amdgcn_global_load_async_to_lds_b128(
      (v4i_glob*)g, (v4i_lds*)l, 0, 0);
#else
  unsigned lds_off = (unsigned)(size_t)l;
  unsigned long long ga = (unsigned long long)(size_t)g;
  asm volatile("global_load_async_to_lds_b128 %0, %1, off"
               :: "v"(lds_off), "v"(ga) : "memory");
#endif
}
__device__ __forceinline__ void wait_async0() {
#if defined(HAVE_WAIT_ASYNC)
  __builtin_amdgcn_s_wait_asynccnt(0);
#else
  asm volatile("s_wait_asynccnt 0x0" ::: "memory");
#endif
}

// ---------------------------------------------------------------------------
// Kernel 1: per-row inclusive prefix sum over T, written TRANSPOSED:
//   csum_t[t * BC + row] = sum(feat[row, 0:t]),  t in [0, T]
// ---------------------------------------------------------------------------
__global__ __launch_bounds__(256) void cumsum_kernel(
    const float* __restrict__ feat, float* __restrict__ csum_t) {
  const int row = blockIdx.x;                 // b*C + c
  const int tid = threadIdx.x;
  const float* in = feat + (size_t)row * Tc;

  float4 a = *(const float4*)(in + tid * 8);
  float4 b = *(const float4*)(in + tid * 8 + 4);
  float x[8] = {a.x, a.y, a.z, a.w, b.x, b.y, b.z, b.w};
  float run = 0.f;
  #pragma unroll
  for (int j = 0; j < 8; ++j) { run += x[j]; x[j] = run; }

  __shared__ float ls[256];
  ls[tid] = run;
  __syncthreads();
  #pragma unroll
  for (int off = 1; off < 256; off <<= 1) {
    float t = (tid >= off) ? ls[tid - off] : 0.f;
    __syncthreads();
    ls[tid] += t;
    __syncthreads();
  }
  const float excl = ls[tid] - run;

  #pragma unroll
  for (int j = 0; j < 8; ++j)
    csum_t[(size_t)(tid * 8 + j + 1) * BCc + row] = excl + x[j];
  if (tid == 0) csum_t[row] = 0.f;
}

// ---------------------------------------------------------------------------
// Kernel 2: SPP pooling -> bf16 activations, row-major [M=B*P][D=7C].
// ---------------------------------------------------------------------------
__global__ __launch_bounds__(256) void spp_pool_kernel(
    const float* __restrict__ csum_t, const int* __restrict__ boxes,
    unsigned short* __restrict__ feats) {
  const int bp = blockIdx.x;                  // b*P + p
  const int b  = bp / Pc;
  int s = boxes[bp * 2 + 0];
  int e = boxes[bp * 2 + 1];
  s = min(max(s, 0), Tc - 1);
  e = max(s + 1, min(e, Tc));
  const int L = e - s;

  int st[7], en[7], obase[7], kstr[7];
  float inv[7];
  int j = 0;
  const int lvls[3]  = {1, 2, 4};
  const int loffs[3] = {0, Cc, 3 * Cc};
  #pragma unroll
  for (int li = 0; li < 3; ++li) {
    const int k = lvls[li];
    for (int i = 0; i < k; ++i) {
      const int a0 = s + (i * L) / k;
      const int a1 = s + ((i + 1) * L + k - 1) / k;
      st[j] = a0; en[j] = a1;
      inv[j] = 1.0f / (float)(a1 - a0);
      obase[j] = loffs[li] + i;
      kstr[j] = k;
      ++j;
    }
  }

  unsigned short* out = feats + (size_t)bp * Dc;
  for (int c = threadIdx.x; c < Cc; c += 256) {
    const int rowidx = b * Cc + c;
    #pragma unroll
    for (int q = 0; q < 7; ++q) {
      const float v = csum_t[(size_t)en[q] * BCc + rowidx] -
                      csum_t[(size_t)st[q] * BCc + rowidx];
      out[obase[q] + c * kstr[q]] = f2bf(v * inv[q]);
    }
  }
}

// ---------------------------------------------------------------------------
// Kernel 3: fp32 -> bf16 weight conversion (vectorized, 8 elems/thread).
// ---------------------------------------------------------------------------
__global__ __launch_bounds__(256) void f32_to_bf16_kernel(
    const float* __restrict__ src, unsigned short* __restrict__ dst) {
  const size_t i = ((size_t)blockIdx.x * 256 + threadIdx.x) * 8;
  float4 a = *(const float4*)(src + i);
  float4 b = *(const float4*)(src + i + 4);
  uint4 pk;
  pk.x = (unsigned)f2bf(a.x) | ((unsigned)f2bf(a.y) << 16);
  pk.y = (unsigned)f2bf(a.z) | ((unsigned)f2bf(a.w) << 16);
  pk.z = (unsigned)f2bf(b.x) | ((unsigned)f2bf(b.y) << 16);
  pk.w = (unsigned)f2bf(b.z) | ((unsigned)f2bf(b.w) << 16);
  *(uint4*)(dst + i) = pk;
}

// ---------------------------------------------------------------------------
// Kernel 4: tiled GEMM  Out[M,N] = relu(A[M,K] * W[N,K]^T + bias)
//   A, W: bf16 row-major K-contiguous.  Core: v_wmma_f32_16x16x32_bf16.
//   Global->LDS via async DMA (no staging VGPRs), double-buffered LDS.
// Block = 256 threads (8 wave32), tile 128x128, K-step 64, LDS 72 KB.
// ---------------------------------------------------------------------------
template <bool OUT_BF16>
__global__ __launch_bounds__(256) void gemm_bf16_nt(
    const unsigned short* __restrict__ A, const unsigned short* __restrict__ W,
    const float* __restrict__ bias, void* __restrict__ Out,
    int M, int N, int K) {
  constexpr int BM = 128, BN = 128, BK = 64, PITCH = 72;  // 144B LDS rows
  __shared__ __align__(16) unsigned short sA[2][BM * PITCH];
  __shared__ __align__(16) unsigned short sB[2][BN * PITCH];

  const int tid  = threadIdx.x;
  const int m0   = blockIdx.y * BM;
  const int n0   = blockIdx.x * BN;
  const int wave = tid >> 5, lane = tid & 31;
  const int mw   = wave >> 2;            // 0..1  -> 64 rows each
  const int nw   = wave & 3;             // 0..3  -> 32 cols each
  const int lr   = lane & 15;
  const int ksel = lane >> 4;            // ISA 16-bit A/B half-K selector

  v8f acc[4][2] = {};

  // Each thread owns 4 16B chunks of the A tile and 4 of the B tile per stage.
  const int row8 = tid >> 3, seg8 = tid & 7;   // chunk = tid + i*256
  auto issue = [&](int kt, int buf) {
    #pragma unroll
    for (int i = 0; i < 4; ++i) {
      const int row = row8 + i * 32;           // (tid + i*256) >> 3
      async_copy16(A + (size_t)(m0 + row) * K + kt + seg8 * 8,
                   &sA[buf][row * PITCH + seg8 * 8]);
    }
    #pragma unroll
    for (int i = 0; i < 4; ++i) {
      const int row = row8 + i * 32;
      async_copy16(W + (size_t)(n0 + row) * K + kt + seg8 * 8,
                   &sB[buf][row * PITCH + seg8 * 8]);
    }
  };

  issue(0, 0);
  const int nStage = K / BK;
  for (int s = 0; s < nStage; ++s) {
    const int buf = s & 1;
    wait_async0();        // this wave's stage-s DMA landed
    __syncthreads();      // all waves' stage-s DMA landed; prev compute done
    if (s + 1 < nStage) issue((s + 1) * BK, buf ^ 1);

    #pragma unroll
    for (int kb = 0; kb < BK; kb += 32) {
      Frag16 af[4], bf[2];
      #pragma unroll
      for (int fm = 0; fm < 4; ++fm) {
        const unsigned short* base =
            &sA[buf][(mw * 64 + fm * 16 + lr) * PITCH + kb + ksel * 8];
        af[fm].h[0] = *(const v8bf*)(base);
        af[fm].h[1] = *(const v8bf*)(base + 16);
      }
      #pragma unroll
      for (int fn = 0; fn < 2; ++fn) {
        const unsigned short* base =
            &sB[buf][(nw * 32 + fn * 16 + lr) * PITCH + kb + ksel * 8];
        bf[fn].h[0] = *(const v8bf*)(base);
        bf[fn].h[1] = *(const v8bf*)(base + 16);
      }
      #pragma unroll
      for (int fm = 0; fm < 4; ++fm)
        #pragma unroll
        for (int fn = 0; fn < 2; ++fn)
          acc[fm][fn] = __builtin_amdgcn_wmma_f32_16x16x32_bf16(
              false, af[fm].v, false, bf[fn].v, (short)0, acc[fm][fn],
              false, false);
    }
  }

  // Epilogue: C/D layout — VGPR i: lane<16 -> M=i, lane>=16 -> M=8+i; N=lane&15
  #pragma unroll
  for (int fm = 0; fm < 4; ++fm) {
    #pragma unroll
    for (int fn = 0; fn < 2; ++fn) {
      const int col = n0 + nw * 32 + fn * 16 + lr;
      const float bv = bias[col];
      #pragma unroll
      for (int i = 0; i < 8; ++i) {
        const int row = m0 + mw * 64 + fm * 16 + ksel * 8 + i;
        float v = acc[fm][fn][i] + bv;
        v = v > 0.f ? v : 0.f;
        if (OUT_BF16)
          ((unsigned short*)Out)[(size_t)row * N + col] = f2bf(v);
        else
          ((float*)Out)[(size_t)row * N + col] = v;
      }
    }
  }
}

// ---------------------------------------------------------------------------
// Kernel 5: head logits. One wave per (row, output); 103 outputs.
// ---------------------------------------------------------------------------
__global__ __launch_bounds__(256) void head_logits_kernel(
    const float* __restrict__ x2,
    const float* __restrict__ cls_w, const float* __restrict__ cls_b,
    const float* __restrict__ det_w, const float* __restrict__ det_b,
    const float* __restrict__ ref_w, const float* __restrict__ ref_b,
    float* __restrict__ cls_log, float* __restrict__ det_log,
    float* __restrict__ ref_log) {
  const int row  = blockIdx.x;
  const int wave = threadIdx.x >> 5, lane = threadIdx.x & 31;
  const int o = blockIdx.y * 8 + wave;
  if (o >= 2 * NCc + Rc * (NCc + 1)) return;       // 103

  const float* w;
  float bv;
  float* dst;
  if (o < NCc) {
    w = cls_w + (size_t)o * Hc; bv = cls_b[o]; dst = cls_log + row * NCc + o;
  } else if (o < 2 * NCc) {
    const int oo = o - NCc;
    w = det_w + (size_t)oo * Hc; bv = det_b[oo]; dst = det_log + row * NCc + oo;
  } else {
    const int oo = o - 2 * NCc;                    // 0..62
    w = ref_w + (size_t)oo * Hc; bv = ref_b[oo];
    dst = ref_log + row * (Rc * (NCc + 1)) + oo;
  }

  const float* xr = x2 + (size_t)row * Hc;
  float s = 0.f;
  for (int h = lane; h < Hc; h += 32) s += xr[h] * w[h];
  s = wred_sum(s);
  if (lane == 0) *dst = s + bv;
}

// ---------------------------------------------------------------------------
// Kernel 6: per-row softmaxes. cls -> cls_sm; refine r -> d_out directly.
// ---------------------------------------------------------------------------
__global__ __launch_bounds__(32) void softmax_rows_kernel(
    const float* __restrict__ cls_log, const float* __restrict__ ref_log,
    float* __restrict__ cls_sm, float* __restrict__ out_refine) {
  const int row = blockIdx.x;
  const int lane = threadIdx.x;

  {
    float v = (lane < NCc) ? cls_log[row * NCc + lane] : -3.0e38f;
    const float mx = wred_max(v);
    float ex = (lane < NCc) ? __expf(v - mx) : 0.f;
    const float sm = wred_sum(ex);
    if (lane < NCc) cls_sm[row * NCc + lane] = ex / sm;
  }
  #pragma unroll
  for (int r = 0; r < Rc; ++r) {
    float v = (lane < NCc + 1)
                  ? ref_log[row * (Rc * (NCc + 1)) + r * (NCc + 1) + lane]
                  : -3.0e38f;
    const float mx = wred_max(v);
    float ex = (lane < NCc + 1) ? __expf(v - mx) : 0.f;
    const float sm = wred_sum(ex);
    if (lane < NCc + 1)
      out_refine[(size_t)r * Mc * (NCc + 1) + row * (NCc + 1) + lane] = ex / sm;
  }
}

// ---------------------------------------------------------------------------
// Kernel 7: det softmax over the proposal axis, then mil_score -> d_out.
// ---------------------------------------------------------------------------
__global__ __launch_bounds__(Pc) void det_softmax_mil_kernel(
    const float* __restrict__ det_log, const float* __restrict__ cls_sm,
    float* __restrict__ out_mil) {
  const int b = blockIdx.x / NCc;
  const int c = blockIdx.x % NCc;
  const int p = threadIdx.x;
  const int row = b * Pc + p;
  const float v = det_log[row * NCc + c];

  __shared__ float red[Pc];
  red[p] = v;
  __syncthreads();
  #pragma unroll
  for (int o = Pc / 2; o > 0; o >>= 1) {
    if (p < o) red[p] = fmaxf(red[p], red[p + o]);
    __syncthreads();
  }
  const float mx = red[0];
  __syncthreads();
  const float ex = __expf(v - mx);
  red[p] = ex;
  __syncthreads();
  #pragma unroll
  for (int o = Pc / 2; o > 0; o >>= 1) {
    if (p < o) red[p] += red[p + o];
    __syncthreads();
  }
  const float s = red[0];
  out_mil[row * NCc + c] = cls_sm[row * NCc + c] * (ex / s);
}

// ---------------------------------------------------------------------------
extern "C" void kernel_launch(void* const* d_in, const int* in_sizes, int n_in,
                              void* d_out, int out_size, void* d_ws,
                              size_t ws_size, hipStream_t stream) {
  (void)in_sizes; (void)n_in; (void)out_size; (void)ws_size;

  const float* feat  = (const float*)d_in[0];
  const int*   boxes = (const int*)d_in[1];
  const float* fc1_w = (const float*)d_in[2];
  const float* fc1_b = (const float*)d_in[3];
  const float* fc2_w = (const float*)d_in[4];
  const float* fc2_b = (const float*)d_in[5];
  const float* cls_w = (const float*)d_in[6];
  const float* cls_b = (const float*)d_in[7];
  const float* det_w = (const float*)d_in[8];
  const float* det_b = (const float*)d_in[9];
  const float* ref_w = (const float*)d_in[10];
  const float* ref_b = (const float*)d_in[11];

  // Workspace carve-up (~206 MB), 256B aligned.
  // Region0 is time-shared: csum (67.1 MB, dead after SPP) then bf16 fc1_w.
  char* ws = (char*)d_ws;
  size_t off = 0;
  auto alloc = [&](size_t bytes) {
    char* p = ws + off;
    off = (off + bytes + 255) & ~(size_t)255;
    return p;
  };
  char* region0 = alloc((size_t)Hc * Dc * 2);                 // 117.4 MB
  float*          csum_t = (float*)region0;                   // phase 1
  unsigned short* w1b    = (unsigned short*)region0;          // phase 2
  unsigned short* feats  = (unsigned short*)alloc((size_t)Mc * Dc * 2);  // 29.4 MB
  unsigned short* w2b    = (unsigned short*)alloc((size_t)Hc * Hc * 2);  // 33.6 MB
  unsigned short* x1     = (unsigned short*)alloc((size_t)Mc * Hc * 2);  //  8.4 MB
  float*          x2     = (float*)alloc((size_t)Mc * Hc * 4);           // 16.8 MB
  float*          cls_log = (float*)alloc((size_t)Mc * NCc * 4);
  float*          det_log = (float*)alloc((size_t)Mc * NCc * 4);
  float*          ref_log = (float*)alloc((size_t)Mc * Rc * (NCc + 1) * 4);
  float*          cls_sm  = (float*)alloc((size_t)Mc * NCc * 4);

  float* out_mil    = (float*)d_out;                          // [B,P,20]
  float* out_refine = (float*)d_out + (size_t)Mc * NCc;       // 3 x [B,P,21]

  // 1) transposed prefix sums (csum lives in region0)
  cumsum_kernel<<<BCc, 256, 0, stream>>>(feat, csum_t);
  // 2) SPP pooling -> bf16 activations (last consumer of csum)
  spp_pool_kernel<<<Mc, 256, 0, stream>>>(csum_t, boxes, feats);
  // 3) weight conversions (w1b overwrites csum, stream-ordered after SPP)
  f32_to_bf16_kernel<<<((size_t)Hc * Dc) / 2048, 256, 0, stream>>>(fc1_w, w1b);
  f32_to_bf16_kernel<<<((size_t)Hc * Hc) / 2048, 256, 0, stream>>>(fc2_w, w2b);
  // 4) fc1: [1024,14336] x [4096,14336]^T, relu -> bf16
  gemm_bf16_nt<true><<<dim3(Hc / 128, Mc / 128), 256, 0, stream>>>(
      feats, w1b, fc1_b, (void*)x1, Mc, Hc, Dc);
  // 5) fc2: [1024,4096] x [4096,4096]^T, relu -> fp32
  gemm_bf16_nt<false><<<dim3(Hc / 128, Mc / 128), 256, 0, stream>>>(
      x1, w2b, fc2_b, (void*)x2, Mc, Hc, Hc);
  // 6) head logits (103 outputs per row)
  head_logits_kernel<<<dim3(Mc, 13), 256, 0, stream>>>(
      x2, cls_w, cls_b, det_w, det_b, ref_w, ref_b, cls_log, det_log, ref_log);
  // 7) class-axis softmaxes (cls + 3 refine heads)
  softmax_rows_kernel<<<Mc, 32, 0, stream>>>(cls_log, ref_log, cls_sm,
                                             out_refine);
  // 8) proposal-axis softmax + mil_score
  det_softmax_mil_kernel<<<Bc * NCc, Pc, 0, stream>>>(det_log, cls_sm, out_mil);
}
